// MultiHeadAttention_80659485819035
// MI455X (gfx1250) — compile-verified, matrix operand order checked by reading, not run on hardware
//
#include <hip/hip_runtime.h>
#include <hip/hip_bf16.h>

// MHA: B=4, H=8, S=2048, depth=64.  Inputs (dict order): v, k, q, mask (all f32).
// Outputs: concat [B,S,512] then attn [B,H,S,S], concatenated flat in d_out.
//
// Memory-bound on the 536MB attn write + 536MB mask read -> single fused pass
// per 16-row q-tile: QK^T (bf16 WMMA, f32 acc, K staged to LDS via async-to-LDS
// copies) -> logits strip in LDS -> softmax in LDS -> write attn once ->
// P@V (bf16 WMMA, V transposed in-flight by per-lane async-to-LDS b32 copies).

#define NB   4
#define NH   8
#define NS   2048
#define ND   64
#define NDM  512
#define NEG_BIG  (-1.0e9f)
#define SCALE    0.125f           // 1/sqrt(64)

#define KSTR 68                   // padded K-row stride in floats (272B, 16B aligned)
#define VSTR 34                   // padded V^T row stride in floats (136B)

typedef __attribute__((ext_vector_type(16))) __bf16 v16bf;
typedef __attribute__((ext_vector_type(8)))  float  v8f;
typedef __attribute__((ext_vector_type(4)))  int    v4i;

__device__ __forceinline__ __bf16 f2bf(float x) { return (__bf16)x; }

#if __has_builtin(__builtin_amdgcn_global_load_async_to_lds_b128) && \
    __has_builtin(__builtin_amdgcn_global_load_async_to_lds_b32)
#define HAVE_ASYNC_LDS 1
#else
#define HAVE_ASYNC_LDS 0
#endif

// addrspace(1)=global ("__device__"), addrspace(3)=LDS ("__shared__") typed casts
#define GAS128(p) ((__attribute__((address_space(1))) v4i*)(p))
#define LAS128(p) ((__attribute__((address_space(3))) v4i*)(p))
#define GAS32(p)  ((__attribute__((address_space(1))) int*)(p))
#define LAS32(p)  ((__attribute__((address_space(3))) int*)(p))

__device__ __forceinline__ void wait_async0() {
#if __has_builtin(__builtin_amdgcn_s_wait_asynccnt)
    __builtin_amdgcn_s_wait_asynccnt(0);
#else
    asm volatile("s_wait_asynccnt 0" ::: "memory");
#endif
}

__global__ __launch_bounds__(256, 1)
void mha_fused_kernel(const float* __restrict__ Vp, const float* __restrict__ Kp,
                      const float* __restrict__ Qp, const float* __restrict__ Mp,
                      float* __restrict__ out_concat, float* __restrict__ out_attn)
{
    extern __shared__ char smem[];
    // LDS carve (~165 KB total; fits 320KB WGP LDS; 1 WG/WGP, BW-bound anyway)
    float*  s_logits = (float*)smem;                         // 16*2048 f32 = 128KB
    __bf16* s_q      = (__bf16*)(smem + 131072);             // 16*64 bf16  = 2KB
    float*  s_part   = (float*)(smem + 133120);              // 16*16 f32
    float*  s_rinv   = (float*)(smem + 134144);              // 16 f32
    float*  s_k      = (float*)(smem + 134208);              // 128*KSTR f32 = 34KB (phase 1)
    float*  s_vt     = (float*)(smem + 134208);              // 64*VSTR f32 (phase 4, aliases s_k)
    float*  s_acc    = (float*)(smem + 134208 + 64*VSTR*4);  // 8*8*32 f32 (phase 4, aliases s_k)

    const int tid  = threadIdx.x;
    const int wave = tid >> 5;
    const int lane = tid & 31;

    const int wg    = blockIdx.x;
    const int qt    = wg & 127;          // q-tile (S/16 = 128)
    const int h     = (wg >> 7) & 7;
    const int b     = wg >> 10;
    const int qbase = qt << 4;

    const int mrow    = lane & 15;       // A-matrix / output row within tile
    const int halfsel = lane >> 4;       // K-subgroup select per ISA layout
    const int ncol16  = lane & 15;       // B/C/D column within tile

    // ---------------- Phase 0: Q tile -> LDS as bf16 ----------------
    for (int i = tid; i < 16 * ND; i += 256) {
        int r = i >> 6, d = i & 63;
        float qv = Qp[((size_t)(b * NS + qbase + r)) * NDM + h * ND + d];
        s_q[r * ND + d] = f2bf(qv);
    }
    __syncthreads();

    // Per-wave A fragments for Q (16x32 bf16, two K-steps for depth 64).
    // ISA 16-bit A layout: lane(0-15)=row M, VGPR j holds K pair at
    //   kk = (j>>2)*16 + halfsel*8 + (j&3)*2
    v16bf a0, a1;
#pragma unroll
    for (int j = 0; j < 8; ++j) {
        int kk = ((j >> 2) << 4) + (halfsel << 3) + ((j & 3) << 1);
        a0[2*j]   = s_q[mrow * ND + kk];
        a0[2*j+1] = s_q[mrow * ND + kk + 1];
        a1[2*j]   = s_q[mrow * ND + 32 + kk];
        a1[2*j+1] = s_q[mrow * ND + 32 + kk + 1];
    }

    // ---------------- Phase 1: logits = QK^T * scale + mask*NEG_BIG ----------------
    // 16 chunks of 128 keys; chunk staged to LDS (async DMA when available),
    // then wave w computes key tile (chunk*8 + w).
    for (int kch = 0; kch < NS / 128; ++kch) {
#if HAVE_ASYNC_LDS
        // 128 rows x 64 floats, natural layout, 16B per lane, fully coalesced.
        for (int i = tid; i < 128 * 16; i += 256) {
            int kk = i >> 4, seg = i & 15;            // seg = 16B segment in row
            const float* g = Kp + ((size_t)(b * NS + kch * 128 + kk)) * NDM + h * ND + seg * 4;
            __builtin_amdgcn_global_load_async_to_lds_b128(
                GAS128(const_cast<float*>(g)), LAS128(s_k + kk * KSTR + seg * 4), 0, 0);
        }
        wait_async0();
#else
        for (int i = tid; i < 128 * ND; i += 256) {
            int kk = i >> 6, d = i & 63;
            s_k[kk * KSTR + d] = Kp[((size_t)(b * NS + kch * 128 + kk)) * NDM + h * ND + d];
        }
#endif
        __syncthreads();

        if (kch + 1 < NS / 128)   // prefetch next mask chunk (global_prefetch_b8)
            __builtin_prefetch(Mp + ((size_t)(b * NS + qbase + mrow)) * NS
                                  + (kch + 1) * 128 + wave * 16, 0, 0);

        // B fragments from LDS K chunk (pairs along depth are contiguous).
        const float* krow = s_k + (wave * 16 + ncol16) * KSTR;
        v16bf b0, b1;
#pragma unroll
        for (int j = 0; j < 8; ++j) {
            int kk = ((j >> 2) << 4) + (halfsel << 3) + ((j & 3) << 1);
            float2 p0 = *(const float2*)(krow + kk);        // ds_load_b64
            float2 p1 = *(const float2*)(krow + 32 + kk);
            b0[2*j] = f2bf(p0.x); b0[2*j+1] = f2bf(p0.y);
            b1[2*j] = f2bf(p1.x); b1[2*j+1] = f2bf(p1.y);
        }
        v8f c = {};
        c = __builtin_amdgcn_wmma_f32_16x16x32_bf16(false, a0, false, b0, (short)0, c, false, false);
        c = __builtin_amdgcn_wmma_f32_16x16x32_bf16(false, a1, false, b1, (short)0, c, false, false);

        const int col = kch * 128 + wave * 16 + ncol16;
#pragma unroll
        for (int r = 0; r < 8; ++r) {
            int m = r + (halfsel << 3);   // C/D layout: lanes16-31 hold M+8
            float mval = Mp[((size_t)(b * NS + qbase + m)) * NS + col];
            s_logits[m * NS + col] = c[r] * SCALE + mval * NEG_BIG;
        }
        __syncthreads();                  // s_k reused next chunk
    }

    // ---------------- Phase 2: row softmax in LDS ----------------
    {
        const int row   = tid >> 4;      // 0..15
        const int cpart = tid & 15;      // 16 threads per row
        float pmax = -INFINITY;
        for (int i = 0; i < 128; ++i)
            pmax = fmaxf(pmax, s_logits[row * NS + cpart * 128 + i]);
        s_part[row * 16 + cpart] = pmax;
        __syncthreads();
        float rmax = -INFINITY;
        for (int i = 0; i < 16; ++i) rmax = fmaxf(rmax, s_part[row * 16 + i]);
        __syncthreads();                 // everyone done reading s_part
        float psum = 0.0f;
        for (int i = 0; i < 128; ++i) {
            int idx = row * NS + cpart * 128 + i;
            float e = __expf(s_logits[idx] - rmax);
            s_logits[idx] = e;
            psum += e;
        }
        s_part[row * 16 + cpart] = psum;
        __syncthreads();
        float rsum = 0.0f;
        for (int i = 0; i < 16; ++i) rsum += s_part[row * 16 + i];
        if (cpart == 0) s_rinv[row] = 1.0f / rsum;
    }
    __syncthreads();

    // ---------------- Phase 3: normalize, write attn (dominant traffic) ----------------
    {
        const size_t attn_base = (((size_t)(b * NH + h)) * NS + qbase) * NS;
        for (int i = tid; i < 16 * NS; i += 256) {
            int m = i >> 11, col = i & 2047;
            float p = s_logits[i] * s_rinv[m];
            s_logits[i] = p;                                  // keep P resident for phase 4
            out_attn[attn_base + (size_t)m * NS + col] = p;   // coalesced 1-time write
        }
    }
    __syncthreads();

    // ---------------- Phase 4: out = P @ V ----------------
    // 8 waves = 4 depth tiles x 2 K-halves (partials combined in LDS).
    // V chunk (32 keys x 64 depth) is TRANSPOSED in flight by per-lane
    // async-to-LDS b32 copies: coalesced global read, [d][key] LDS layout.
    const int nt    = wave & 3;          // depth tile: cols nt*16..nt*16+15
    const int khalf = wave >> 2;         // even/odd key chunks
    const int dcol  = (nt << 4) + ncol16;
    v8f co = {};
    for (int kc = 0; kc < NS / 32; ++kc) {
#if HAVE_ASYNC_LDS
        for (int i = tid; i < 32 * ND; i += 256) {
            int kk = i >> 6, d = i & 63;             // consecutive lanes: consecutive d
            const float* g = Vp + ((size_t)(b * NS + kc * 32 + kk)) * NDM + h * ND + d;
            __builtin_amdgcn_global_load_async_to_lds_b32(
                GAS32(const_cast<float*>(g)), LAS32(s_vt + d * VSTR + kk), 0, 0);
        }
        wait_async0();
#else
        for (int i = tid; i < 32 * ND; i += 256) {
            int kk = i >> 6, d = i & 63;
            s_vt[d * VSTR + kk] = Vp[((size_t)(b * NS + kc * 32 + kk)) * NDM + h * ND + d];
        }
#endif
        __syncthreads();
        if ((kc & 1) == khalf) {
            v16bf pa, vb;
#pragma unroll
            for (int j = 0; j < 8; ++j) {
                int kk = ((j >> 2) << 4) + (halfsel << 3) + ((j & 3) << 1);
                int kg = kc * 32 + kk;
                pa[2*j]   = f2bf(s_logits[mrow * NS + kg]);
                pa[2*j+1] = f2bf(s_logits[mrow * NS + kg + 1]);
                float2 vv = *(const float2*)(s_vt + dcol * VSTR + kk);  // ds_load_b64
                vb[2*j]   = f2bf(vv.x);
                vb[2*j+1] = f2bf(vv.y);
            }
            co = __builtin_amdgcn_wmma_f32_16x16x32_bf16(false, pa, false, vb, (short)0, co, false, false);
        }
        __syncthreads();                 // s_vt reused next chunk
    }
#pragma unroll
    for (int r = 0; r < 8; ++r)
        s_acc[(wave * 8 + r) * 32 + lane] = co[r];
    __syncthreads();

    // combine K-half partials and write concat output
    for (int i = tid; i < 1024; i += 256) {
        int w0 = i >> 8;                 // depth tile 0..3
        int rr = (i >> 5) & 7;           // C vgpr index
        int ln = i & 31;
        float val = s_acc[(w0 * 8 + rr) * 32 + ln] + s_acc[((w0 + 4) * 8 + rr) * 32 + ln];
        int m  = rr + ((ln >> 4) << 3);
        int dc = (w0 << 4) + (ln & 15);
        out_concat[((size_t)(b * NS + qbase + m)) * NDM + h * ND + dc] = val;
    }
}

extern "C" void kernel_launch(void* const* d_in, const int* in_sizes, int n_in,
                              void* d_out, int out_size, void* d_ws, size_t ws_size,
                              hipStream_t stream) {
    const float* Vp = (const float*)d_in[0];
    const float* Kp = (const float*)d_in[1];
    const float* Qp = (const float*)d_in[2];
    const float* Mp = (const float*)d_in[3];
    float* out_concat = (float*)d_out;
    float* out_attn   = out_concat + (size_t)NB * NS * NDM;

    const size_t smem = 134208                      // logits + q + reductions
                      + (size_t)128 * KSTR * 4;     // K chunk (aliased by V^T + acc)

    (void)hipFuncSetAttribute(reinterpret_cast<const void*>(mha_fused_kernel),
                              hipFuncAttributeMaxDynamicSharedMemorySize, (int)smem);

    dim3 grid(NB * NH * (NS / 16));
    mha_fused_kernel<<<grid, 256, smem, stream>>>(Vp, Kp, Qp, Mp, out_concat, out_attn);
}
